// BasicGNN_16733192585664
// MI455X (gfx1250) — matrix operand order repaired and checked by reading, not compiled, not run on hardware
//
#include <hip/hip_runtime.h>
#include <hip/hip_bf16.h>
#include <math.h>

// ---------------- types ----------------
typedef __attribute__((ext_vector_type(16))) __bf16 v16bf;
typedef __attribute__((ext_vector_type(8)))  float  v8f;

union Frag16 { uint4 u[2]; v16bf v; };

static __device__ __forceinline__ __bf16 f2bf(float f) {
    union { float f; unsigned u; } in; in.f = f;
    unsigned r = in.u + 0x7FFFu + ((in.u >> 16) & 1u);   // RNE
    unsigned short h = (unsigned short)(r >> 16);
    union { unsigned short s; __bf16 b; } out; out.s = h;
    return out.b;
}

#define H 64

// Stage 8192 bf16 (16KB) of packed B-fragments from global into LDS using the
// CDNA5 async-to-LDS path (ASYNCcnt). blockDim.x must be 128.
static __device__ __forceinline__ void stage_wB_to_lds(const __bf16* __restrict__ wB,
                                                       __bf16* __restrict__ sWB) {
    #pragma unroll
    for (int i = 0; i < 8; ++i) {
        int idx = ((int)threadIdx.x + i * 128) * 8;          // 8 halves = 16B per chunk
        unsigned lds_off = (unsigned)(size_t)(&sWB[idx]);    // LDS offset = low 32 bits
        const __bf16* g = wB + idx;
        asm volatile("global_load_async_to_lds_b128 %0, %1, off"
                     :: "v"(lds_off), "v"(g)
                     : "memory");
    }
    asm volatile("s_wait_asynccnt 0x0" ::: "memory");
    __syncthreads();
}

// ---------------- small utility kernels ----------------
__global__ void zero_f32_kernel(float* __restrict__ p, int n) {
    int i = blockIdx.x * blockDim.x + threadIdx.x;
    if (i < n) p[i] = 0.0f;
}

// Stack two 64x64 (or a contiguous 128x64) f32 weight(s) into pre-swizzled
// bf16 B-fragments for v_wmma_f32_16x16x32_bf16.
// Fragment order: idx = ((kb*4+nt)*32 + lane)*16 + j
//   g = lane>>4, n = nt*16 + (lane&15), k = kb*32 + (j<8 ? g*8 + j : 16 + g*8 + (j-8))
__global__ void pack_b_kernel(const float* __restrict__ Wl,   // rows 0..63   [64x64]
                              const float* __restrict__ Wr,   // rows 64..127 [64x64]
                              __bf16* __restrict__ out) {     // 8192 halves
    int idx = blockIdx.x * blockDim.x + threadIdx.x;
    if (idx >= 8192) return;
    int j    = idx & 15;
    int lane = (idx >> 4) & 31;
    int frag = idx >> 9;          // 0..15
    int nt = frag & 3;
    int kb = frag >> 2;
    int g  = lane >> 4;
    int n  = nt * 16 + (lane & 15);
    int k  = kb * 32 + ((j < 8) ? (g * 8 + j) : (16 + g * 8 + (j - 8)));
    float v = (k < 64) ? Wl[k * H + n] : Wr[(k - 64) * H + n];
    out[idx] = f2bf(v);
}

// h = x @ W + b   (x:[N,5], W:[5,64], b:[64])
__global__ void proj_kernel(const float* __restrict__ x, const float* __restrict__ W,
                            const float* __restrict__ b, float* __restrict__ h, int N) {
    int i = blockIdx.x * blockDim.x + threadIdx.x;
    if (i >= N * H) return;
    int n = i >> 6, j = i & 63;
    const float* xr = x + (size_t)n * 5;
    float acc = b[j];
    #pragma unroll
    for (int k = 0; k < 5; ++k) acc += xr[k] * W[k * H + j];
    h[i] = acc;
}

// degrees: deg_mft counts over dst, deg_mch counts over src
__global__ void degree_kernel(const int* __restrict__ src, const int* __restrict__ dst,
                              float* __restrict__ deg_mch, float* __restrict__ deg_mft, int E) {
    int e = blockIdx.x * blockDim.x + threadIdx.x;
    if (e >= E) return;
    atomicAdd(deg_mch + src[e], 1.0f);
    atomicAdd(deg_mft + dst[e], 1.0f);
}

// scatter both directions: feat_mch[src] -> agg_mft[dst], feat_mft[dst] -> agg_mch[src]
__global__ void scatter_kernel(const float* __restrict__ feat_mch,
                               const float* __restrict__ feat_mft,
                               const int* __restrict__ src, const int* __restrict__ dst,
                               float* __restrict__ agg_mft, float* __restrict__ agg_mch, int E) {
    int tid = blockIdx.x * blockDim.x + threadIdx.x;
    int e = tid >> 4;
    if (e >= E) return;
    int seg = (tid & 15) * 4;
    int s = src[e], d = dst[e];
    float4 a = *(const float4*)(const void*)(feat_mch + (size_t)s * H + seg);
    float4 b = *(const float4*)(const void*)(feat_mft + (size_t)d * H + seg);
    float* pa = agg_mft + (size_t)d * H + seg;
    float* pb = agg_mch + (size_t)s * H + seg;
    atomicAdd(pa + 0, a.x); atomicAdd(pa + 1, a.y);
    atomicAdd(pa + 2, a.z); atomicAdd(pa + 3, a.w);
    atomicAdd(pb + 0, b.x); atomicAdd(pb + 1, b.y);
    atomicAdd(pb + 2, b.z); atomicAdd(pb + 3, b.w);
}

// cat[n] = [ bf16(agg[n]/max(deg[n],1)) , bf16(h[n]) ]  -> [N,128] bf16
__global__ void prep_cat_kernel(const float* __restrict__ agg, const float* __restrict__ deg,
                                const float* __restrict__ h, __bf16* __restrict__ cat, int N) {
    int i = blockIdx.x * blockDim.x + threadIdx.x;
    if (i >= N * H) return;
    int n = i >> 6, j = i & 63;
    float dv = fmaxf(deg[n], 1.0f);
    cat[(size_t)n * 128 + j]      = f2bf(agg[i] / dv);
    cat[(size_t)n * 128 + 64 + j] = f2bf(h[i]);
}

// ---------------- WMMA node-update GEMM ----------------
// out = cat[N,128] @ Wcat[128,64] + bias ; mode 0: relu -> f32, mode 1: plain -> bf16
__global__ void node_gemm_kernel(const __bf16* __restrict__ cat,
                                 const __bf16* __restrict__ wB,   // packed frags (8192 halves)
                                 const float* __restrict__ bias,
                                 float* __restrict__ outF,
                                 __bf16* __restrict__ outB,
                                 int N, int mode) {
    __shared__ __bf16 sWB[8192];
    stage_wB_to_lds(wB, sWB);                    // async-to-LDS + barrier (all waves)

    int lane = threadIdx.x & 31;
    int wave = blockIdx.x * (blockDim.x >> 5) + (threadIdx.x >> 5);
    int rowBase = wave * 16;
    if (rowBase >= N) return;                    // wave-uniform exit (after barrier)
    int g = lane >> 4;
    int row = rowBase + (lane & 15);
    int rowC = row < N ? row : N - 1;            // keep EXEC full for WMMA
    const __bf16* aRow = cat + (size_t)rowC * 128;

    v8f acc[4];
    #pragma unroll
    for (int t = 0; t < 4; ++t)
        #pragma unroll
        for (int r = 0; r < 8; ++r) acc[t][r] = 0.0f;

    #pragma unroll
    for (int kb = 0; kb < 4; ++kb) {
        Frag16 a;
        a.u[0] = *(const uint4*)(const void*)(aRow + kb * 32 + g * 8);
        a.u[1] = *(const uint4*)(const void*)(aRow + kb * 32 + 16 + g * 8);
        #pragma unroll
        for (int nt = 0; nt < 4; ++nt) {
            Frag16 b;
            const __bf16* bp = sWB + ((size_t)((kb * 4 + nt) * 32 + lane)) * 16;
            b.u[0] = *(const uint4*)(const void*)(bp);
            b.u[1] = *(const uint4*)(const void*)(bp + 8);
            acc[nt] = __builtin_amdgcn_wmma_f32_16x16x32_bf16(
                false, a.v, false, b.v, (short)0, acc[nt], false, false);
        }
    }
    #pragma unroll
    for (int nt = 0; nt < 4; ++nt) {
        int n = nt * 16 + (lane & 15);
        float bv = bias[n];
        #pragma unroll
        for (int r = 0; r < 8; ++r) {
            int orow = rowBase + r + 8 * g;
            if (orow < N) {
                float v = acc[nt][r] + bv;
                if (mode == 0) outF[(size_t)orow * H + n] = fmaxf(v, 0.0f);
                else           outB[(size_t)orow * H + n] = f2bf(v);
            }
        }
    }
}

// ---------------- fused edge classifier ----------------
// A row e = [mch2b[src[e]] (K 0..63) , mft2b[dst[e]] (K 64..127)]
// h = relu(A @ W1 + b1); out[e] = sigmoid(h . w2 + b2)
__global__ void edge_mlp_kernel(const __bf16* __restrict__ mch2b,
                                const __bf16* __restrict__ mft2b,
                                const int* __restrict__ src, const int* __restrict__ dst,
                                const __bf16* __restrict__ wB,   // packed ec_W1 frags
                                const float* __restrict__ b1,
                                const float* __restrict__ w2, const float* __restrict__ b2,
                                float* __restrict__ out, int E) {
    __shared__ __bf16 sWB[8192];
    stage_wB_to_lds(wB, sWB);                    // async-to-LDS + barrier (all waves)

    int lane = threadIdx.x & 31;
    int wave = blockIdx.x * (blockDim.x >> 5) + (threadIdx.x >> 5);
    int eBase = wave * 16;
    if (eBase >= E) return;                      // wave-uniform exit (after barrier)
    int g = lane >> 4;
    int e = eBase + (lane & 15);
    int eC = e < E ? e : E - 1;                  // keep EXEC full
    int s = src[eC], d = dst[eC];
    const __bf16* rowM = mch2b + (size_t)s * H;
    const __bf16* rowF = mft2b + (size_t)d * H;

    v8f acc[4];
    #pragma unroll
    for (int t = 0; t < 4; ++t)
        #pragma unroll
        for (int r = 0; r < 8; ++r) acc[t][r] = 0.0f;

    #pragma unroll
    for (int kb = 0; kb < 4; ++kb) {
        const __bf16* base = (kb < 2) ? (rowM + kb * 32) : (rowF + (kb - 2) * 32);
        Frag16 a;
        a.u[0] = *(const uint4*)(const void*)(base + g * 8);
        a.u[1] = *(const uint4*)(const void*)(base + 16 + g * 8);
        #pragma unroll
        for (int nt = 0; nt < 4; ++nt) {
            Frag16 b;
            const __bf16* bp = sWB + ((size_t)((kb * 4 + nt) * 32 + lane)) * 16;
            b.u[0] = *(const uint4*)(const void*)(bp);
            b.u[1] = *(const uint4*)(const void*)(bp + 8);
            acc[nt] = __builtin_amdgcn_wmma_f32_16x16x32_bf16(
                false, a.v, false, b.v, (short)0, acc[nt], false, false);
        }
    }

    // second layer: per-lane column n = nt*16 + (lane&15)
    float biasv[4], w2v[4];
    #pragma unroll
    for (int nt = 0; nt < 4; ++nt) {
        int n = nt * 16 + (lane & 15);
        biasv[nt] = b1[n];
        w2v[nt]   = w2[n];
    }
    float bias2 = b2[0];
    #pragma unroll
    for (int r = 0; r < 8; ++r) {
        float sum = 0.0f;
        #pragma unroll
        for (int nt = 0; nt < 4; ++nt) {
            float hv = fmaxf(acc[nt][r] + biasv[nt], 0.0f);
            sum += hv * w2v[nt];
        }
        // reduce across 16 lanes of each half-wave
        #pragma unroll
        for (int off = 8; off >= 1; off >>= 1)
            sum += __shfl_xor(sum, off, 16);
        if ((lane & 15) == 0) {
            int eo = eBase + r + 8 * g;
            if (eo < E) {
                float z = sum + bias2;
                out[eo] = 1.0f / (1.0f + expf(-z));
            }
        }
    }
}

// ---------------- host side ----------------
static inline size_t alignUp(size_t x, size_t a) { return (x + a - 1) & ~(a - 1); }

extern "C" void kernel_launch(void* const* d_in, const int* in_sizes, int n_in,
                              void* d_out, int out_size, void* d_ws, size_t ws_size,
                              hipStream_t stream) {
    const float* x_mch = (const float*)d_in[0];
    const float* x_mft = (const float*)d_in[1];
    const int*   eidx  = (const int*)d_in[2];
    const float* W_mch = (const float*)d_in[3];
    const float* b_mch = (const float*)d_in[4];
    const float* W_mft = (const float*)d_in[5];
    const float* b_mft = (const float*)d_in[6];
    const float* c1m_Wl = (const float*)d_in[7];
    const float* c1m_bl = (const float*)d_in[8];
    const float* c1m_Wr = (const float*)d_in[9];
    const float* c1r_Wl = (const float*)d_in[10];
    const float* c1r_bl = (const float*)d_in[11];
    const float* c1r_Wr = (const float*)d_in[12];
    const float* c2m_Wl = (const float*)d_in[13];
    const float* c2m_bl = (const float*)d_in[14];
    const float* c2m_Wr = (const float*)d_in[15];
    const float* c2r_Wl = (const float*)d_in[16];
    const float* c2r_bl = (const float*)d_in[17];
    const float* c2r_Wr = (const float*)d_in[18];
    const float* ec_W1 = (const float*)d_in[19];
    const float* ec_b1 = (const float*)d_in[20];
    const float* ec_W2 = (const float*)d_in[21];
    const float* ec_b2 = (const float*)d_in[22];

    const int Nmch = in_sizes[0] / 5;
    const int Nmft = in_sizes[1] / 5;
    const int E    = in_sizes[2] / 2;
    const int* src = eidx;
    const int* dst = eidx + E;

    // workspace carve-up
    char* p = (char*)d_ws;
    auto take = [&](size_t bytes) { char* r = p; p += alignUp(bytes, 256); return r; };
    __bf16* wB1m = (__bf16*)take(8192 * sizeof(__bf16));
    __bf16* wB1r = (__bf16*)take(8192 * sizeof(__bf16));
    __bf16* wB2m = (__bf16*)take(8192 * sizeof(__bf16));
    __bf16* wB2r = (__bf16*)take(8192 * sizeof(__bf16));
    __bf16* wBec = (__bf16*)take(8192 * sizeof(__bf16));
    float* h_mch = (float*)take((size_t)Nmch * H * sizeof(float));
    float* h_mft = (float*)take((size_t)Nmft * H * sizeof(float));
    float* deg_mch = (float*)take((size_t)Nmch * sizeof(float));
    float* deg_mft = (float*)take((size_t)Nmft * sizeof(float));
    float* agg_mft = (float*)take((size_t)Nmft * H * sizeof(float));   // neighbors of mft
    float* agg_mch = (float*)take((size_t)Nmch * H * sizeof(float));   // neighbors of mch
    __bf16* cat_mft = (__bf16*)take((size_t)Nmft * 128 * sizeof(__bf16));
    __bf16* cat_mch = (__bf16*)take((size_t)Nmch * 128 * sizeof(__bf16));
    float* o1_mch = (float*)take((size_t)Nmch * H * sizeof(float));
    float* o1_mft = (float*)take((size_t)Nmft * H * sizeof(float));
    __bf16* o2_mch = (__bf16*)take((size_t)Nmch * H * sizeof(__bf16));
    __bf16* o2_mft = (__bf16*)take((size_t)Nmft * H * sizeof(__bf16));
    (void)ws_size; (void)n_in; (void)out_size;

    const int B = 256;
    auto g1 = [&](long n) { return (unsigned)((n + B - 1) / B); };

    // 1) pack weights into bf16 B-fragments
    pack_b_kernel<<<g1(8192), B, 0, stream>>>(c1m_Wl, c1m_Wr, wB1m);
    pack_b_kernel<<<g1(8192), B, 0, stream>>>(c1r_Wl, c1r_Wr, wB1r);
    pack_b_kernel<<<g1(8192), B, 0, stream>>>(c2m_Wl, c2m_Wr, wB2m);
    pack_b_kernel<<<g1(8192), B, 0, stream>>>(c2r_Wl, c2r_Wr, wB2r);
    pack_b_kernel<<<g1(8192), B, 0, stream>>>(ec_W1, ec_W1 + 64 * H, wBec);

    // 2) input projections
    proj_kernel<<<g1((long)Nmch * H), B, 0, stream>>>(x_mch, W_mch, b_mch, h_mch, Nmch);
    proj_kernel<<<g1((long)Nmft * H), B, 0, stream>>>(x_mft, W_mft, b_mft, h_mft, Nmft);

    // 3) degrees (shared by both conv layers)
    zero_f32_kernel<<<g1(Nmch), B, 0, stream>>>(deg_mch, Nmch);
    zero_f32_kernel<<<g1(Nmft), B, 0, stream>>>(deg_mft, Nmft);
    degree_kernel<<<g1(E), B, 0, stream>>>(src, dst, deg_mch, deg_mft, E);

    // ---- conv1 ----
    zero_f32_kernel<<<g1((long)Nmft * H), B, 0, stream>>>(agg_mft, Nmft * H);
    zero_f32_kernel<<<g1((long)Nmch * H), B, 0, stream>>>(agg_mch, Nmch * H);
    scatter_kernel<<<g1((long)E * 16), B, 0, stream>>>(h_mch, h_mft, src, dst, agg_mft, agg_mch, E);
    prep_cat_kernel<<<g1((long)Nmft * H), B, 0, stream>>>(agg_mft, deg_mft, h_mft, cat_mft, Nmft);
    prep_cat_kernel<<<g1((long)Nmch * H), B, 0, stream>>>(agg_mch, deg_mch, h_mch, cat_mch, Nmch);
    {
        int wavesF = (Nmft + 15) / 16, wavesC = (Nmch + 15) / 16;
        node_gemm_kernel<<<(wavesF + 3) / 4, 128, 0, stream>>>(cat_mft, wB1m, c1m_bl, o1_mft, nullptr, Nmft, 0);
        node_gemm_kernel<<<(wavesC + 3) / 4, 128, 0, stream>>>(cat_mch, wB1r, c1r_bl, o1_mch, nullptr, Nmch, 0);
    }

    // ---- conv2 ----
    zero_f32_kernel<<<g1((long)Nmft * H), B, 0, stream>>>(agg_mft, Nmft * H);
    zero_f32_kernel<<<g1((long)Nmch * H), B, 0, stream>>>(agg_mch, Nmch * H);
    scatter_kernel<<<g1((long)E * 16), B, 0, stream>>>(o1_mch, o1_mft, src, dst, agg_mft, agg_mch, E);
    prep_cat_kernel<<<g1((long)Nmft * H), B, 0, stream>>>(agg_mft, deg_mft, o1_mft, cat_mft, Nmft);
    prep_cat_kernel<<<g1((long)Nmch * H), B, 0, stream>>>(agg_mch, deg_mch, o1_mch, cat_mch, Nmch);
    {
        int wavesF = (Nmft + 15) / 16, wavesC = (Nmch + 15) / 16;
        node_gemm_kernel<<<(wavesF + 3) / 4, 128, 0, stream>>>(cat_mft, wB2m, c2m_bl, nullptr, o2_mft, Nmft, 1);
        node_gemm_kernel<<<(wavesC + 3) / 4, 128, 0, stream>>>(cat_mch, wB2r, c2r_bl, nullptr, o2_mch, Nmch, 1);
    }

    // ---- edge classifier (fused gather + WMMA GEMM + dot + sigmoid) ----
    {
        int waves = (E + 15) / 16;
        edge_mlp_kernel<<<(waves + 3) / 4, 128, 0, stream>>>(
            o2_mch, o2_mft, src, dst, wBec, ec_b1, ec_W2, ec_b2, (float*)d_out, E);
    }
}
